// HDTForSummarization_7421703488081
// MI455X (gfx1250) — matrix operand order repaired and checked by reading, not compiled, not run on hardware
//
#include <hip/hip_runtime.h>
#include <hip/hip_bf16.h>
#include <stdint.h>

// ---------------- model constants ----------------
#define KB_ 4          // batch
#define KSRC 1024
#define KTGT 256
#define KD 512         // d_model
#define KH 8           // heads
#define KHD 64         // head dim
#define KV 30522       // vocab
#define KFF 2048       // 4*D
#define KASENT 64      // anchors

typedef __bf16 bf16_t;
typedef bf16_t v16bf __attribute__((ext_vector_type(16)));
typedef float  v8f   __attribute__((ext_vector_type(8)));

__device__ __forceinline__ bf16_t f2bf(float f) {
    unsigned int u = __builtin_bit_cast(unsigned int, f);
    unsigned int r = (u + 0x7FFFu + ((u >> 16) & 1u)) >> 16;
    return __builtin_bit_cast(bf16_t, (unsigned short)r);
}
// pack two floats into one dword of bf16 (lo, hi), round-to-nearest-even
__device__ __forceinline__ unsigned pk2(float a, float b) {
    unsigned ua = __builtin_bit_cast(unsigned, a);
    unsigned ub = __builtin_bit_cast(unsigned, b);
    unsigned ra = (ua + 0x7FFFu + ((ua >> 16) & 1u)) >> 16;
    unsigned rb = (ub + 0x7FFFu + ((ub >> 16) & 1u)) >> 16;
    return ra | (rb << 16);
}

__device__ __forceinline__ float gelu_f(float x) {
    float x3 = x * x * x;
    return 0.5f * x * (1.f + tanhf(0.7978845608f * (x + 0.044715f * x3)));
}

// =====================================================================
// Weight prep: Wt[n*K + k] = bf16(W[k*N + n])   (convert + transpose once)
// =====================================================================
__global__ __launch_bounds__(256) void convT_k(
    const float* __restrict__ W, bf16_t* __restrict__ Wt, int K, int N)
{
    __shared__ float tile[32][33];
    const int k0 = blockIdx.y * 32, n0 = blockIdx.x * 32;
    const int tx = threadIdx.x & 31, ty = threadIdx.x >> 5;   // 32x8
    #pragma unroll
    for (int i = ty; i < 32; i += 8) {
        int k = k0 + i, n = n0 + tx;
        tile[i][tx] = (k < K && n < N) ? W[(size_t)k * N + n] : 0.f;
    }
    __syncthreads();
    #pragma unroll
    for (int i = ty; i < 32; i += 8) {
        int n = n0 + i, k = k0 + tx;
        if (n < N && k < K) Wt[(size_t)n * K + k] = f2bf(tile[tx][i]);
    }
}

// =====================================================================
// GEMM: C[M,N] = epi( (A (+A2)) @ W[K,N] + bias ) (+Res)
// A: f32 [M,K]; weights pre-converted bf16, transposed Wt[N,K].
// Block = 256 threads = 8 wave32; block tile 64x64; wave tile 16x32; K-step 32.
// B tile is staged with GLOBAL_LOAD_ASYNC_TO_LDS_B128 (ASYNCcnt).
// =====================================================================
__global__ __launch_bounds__(256) void gemm_bf16_wmma(
    const float* __restrict__ A, const float* __restrict__ A2,
    const bf16_t* __restrict__ Wt, const float* __restrict__ bias,
    const float* __restrict__ Res, float* __restrict__ C,
    int M, int N, int K, int act)
{
    __shared__ bf16_t As[64][40];   // [m][k], 80B pitch (16B aligned rows)
    __shared__ bf16_t Bs[64][40];   // [n][k]

    const int tid  = threadIdx.x;
    const int lane = tid & 31;
    const int wv   = tid >> 5;
    const int wm   = wv & 3;        // 4 waves along M
    const int wn   = wv >> 2;       // 2 waves along N
    const int n0   = blockIdx.x * 64;
    const int m0   = blockIdx.y * 64;

    v8f acc0 = {0.f,0.f,0.f,0.f,0.f,0.f,0.f,0.f};
    v8f acc1 = {0.f,0.f,0.f,0.f,0.f,0.f,0.f,0.f};

    // A staging: thread -> (row 0..63, kbase 0/8/16/24), 8 floats -> 8 bf16
    const int alr = tid >> 2;
    const int alc = (tid & 3) << 3;
    // B staging: thread -> (n 0..63, kbase 0/8/16/24), 8 bf16 = 16 bytes
    const int bkb = (tid >> 6) << 3;
    const int bnn = tid & 63;

    const int gmA = m0 + alr;
    const int gnB = n0 + bnn;
    const float*  apA  = A + (size_t)gmA * K + alc;
    const float*  apA2 = A2 ? (A2 + (size_t)gmA * K + alc) : nullptr;
    const bf16_t* bp   = Wt + (size_t)gnB * K + bkb;
    // low 32 bits of a generic LDS address are the LDS byte offset
    const unsigned ldsB = (unsigned)(uintptr_t)&Bs[bnn][bkb];

    for (int kt = 0; kt < K; kt += 32) {
        // ---- stage A tile: float4 x2 load, pack, one b128 LDS store ----
        float4 va0 = make_float4(0.f, 0.f, 0.f, 0.f);
        float4 va1 = va0;
        if (gmA < M) {
            const float4* p = (const float4*)(apA + kt);
            va0 = p[0]; va1 = p[1];
            if (apA2) {
                const float4* q = (const float4*)(apA2 + kt);
                float4 w0 = q[0], w1 = q[1];
                va0.x += w0.x; va0.y += w0.y; va0.z += w0.z; va0.w += w0.w;
                va1.x += w1.x; va1.y += w1.y; va1.z += w1.z; va1.w += w1.w;
            }
            if (kt + 32 < K)
                __builtin_prefetch(apA + kt + 32, 0, 3);   // global_prefetch_b8
        }
        uint4 pk;
        pk.x = pk2(va0.x, va0.y); pk.y = pk2(va0.z, va0.w);
        pk.z = pk2(va1.x, va1.y); pk.w = pk2(va1.z, va1.w);
        *(uint4*)&As[alr][alc] = pk;

        // ---- stage B tile: async 16B copy global(bf16,K-major) -> LDS ----
        if (gnB < N) {
            const bf16_t* g = bp + kt;
            asm volatile("global_load_async_to_lds_b128 %0, %1, off"
                         :: "v"(ldsB), "v"(g) : "memory");
        } else {
            uint4 z = {0u, 0u, 0u, 0u};
            *(uint4*)&Bs[bnn][bkb] = z;
        }
        asm volatile("s_wait_asynccnt 0" ::: "memory");
        __syncthreads();

        // ---- fragments in contiguous k order (k = 16*(lane/16) + s) for A and B ----
        const int arow = wm * 16 + (lane & 15);
        const int kbA  = (lane >> 4) << 4;
        v16bf af;
        #pragma unroll
        for (int j = 0; j < 8; ++j) af[j]     = As[arow][kbA + j];
        #pragma unroll
        for (int j = 0; j < 8; ++j) af[8 + j] = As[arow][kbA + 8 + j];

        const int kbB = (lane >> 4) << 4;
        const int bn0 = wn * 32 + (lane & 15);
        v16bf bf0, bf1;
        #pragma unroll
        for (int j = 0; j < 16; ++j) bf0[j] = Bs[bn0][kbB + j];
        #pragma unroll
        for (int j = 0; j < 16; ++j) bf1[j] = Bs[bn0 + 16][kbB + j];

        acc0 = __builtin_amdgcn_wmma_f32_16x16x32_bf16(
                   false, af, false, bf0, (short)0, acc0, false, false);
        acc1 = __builtin_amdgcn_wmma_f32_16x16x32_bf16(
                   false, af, false, bf1, (short)0, acc1, false, false);
        __syncthreads();
    }

    // ---- epilogue: C/D layout — VGPR i: row = i + 8*(lane>=16); col = lane&15
    const int rb = wm * 16 + ((lane >> 4) << 3);
    const int cb = lane & 15;
    #pragma unroll
    for (int i = 0; i < 8; ++i) {
        int gm = m0 + rb + i;
        if (gm >= M) continue;
        int gn = n0 + wn * 32 + cb;
        if (gn < N) {
            float v = acc0[i];
            if (bias) v += bias[gn];
            if (act)  v  = gelu_f(v);
            if (Res)  v += Res[(size_t)gm * N + gn];
            C[(size_t)gm * N + gn] = v;
        }
        gn += 16;
        if (gn < N) {
            float v = acc1[i];
            if (bias) v += bias[gn];
            if (act)  v  = gelu_f(v);
            if (Res)  v += Res[(size_t)gm * N + gn];
            C[(size_t)gm * N + gn] = v;
        }
    }
}

// =====================================================================
// Flash-style attention: one wave32 per (b, head, query); online softmax.
// =====================================================================
__global__ __launch_bounds__(256) void attn_wave(
    const float* __restrict__ Q, const float* __restrict__ Km,
    const float* __restrict__ Vm, float* __restrict__ O,
    const int* __restrict__ mask,
    int nb, int qlen, int klen, int kv_bstride, int window, int causal)
{
    __shared__ float sQ[8][64];
    __shared__ float sP[8][32];
    const int w    = threadIdx.x >> 5;
    const int lane = threadIdx.x & 31;
    const int task = blockIdx.x * 8 + w;
    const int total = nb * KH * qlen;
    if (task >= total) return;
    const int qi = task % qlen;
    const int h  = (task / qlen) % KH;
    const int b  = task / (qlen * KH);

    const float* qrow = Q + ((size_t)(b * qlen + qi)) * KD + h * KHD;
    sQ[w][lane]      = qrow[lane];
    sQ[w][lane + 32] = qrow[lane + 32];

    int lo = 0, hi = klen - 1;
    if (window >= 0) { lo = max(0, qi - window); hi = min(hi, qi + window); }
    if (causal) hi = min(hi, qi);
    const size_t kvbase = (size_t)b * kv_bstride;

    float mrun = -1e30f, lrun = 0.f, acc0 = 0.f, acc1 = 0.f;
    for (int j0 = lo; j0 <= hi; j0 += 32) {
        int j = j0 + lane;
        float s = -1e30f;
        if (j <= hi) {
            const float* krow = Km + (kvbase + j) * KD + h * KHD;
            float d = 0.f;
            #pragma unroll
            for (int t = 0; t < KHD; ++t) d += sQ[w][t] * krow[t];
            s = d * 0.125f;                       // 1/sqrt(64)
            if (mask && mask[b * klen + j] == 0) s = -10000.0f;
        }
        float cm = s;
        #pragma unroll
        for (int o = 16; o > 0; o >>= 1) cm = fmaxf(cm, __shfl_xor(cm, o, 32));
        float mnew = fmaxf(mrun, cm);
        float p = (j <= hi) ? __expf(s - mnew) : 0.f;
        float ps = p;
        #pragma unroll
        for (int o = 16; o > 0; o >>= 1) ps += __shfl_xor(ps, o, 32);
        float corr = __expf(mrun - mnew);
        lrun = lrun * corr + ps;
        acc0 *= corr; acc1 *= corr;
        mrun = mnew;
        sP[w][lane] = p;
        int nval = min(32, hi - j0 + 1);
        for (int jj = 0; jj < nval; ++jj) {
            float pj = sP[w][jj];
            const float* vrow = Vm + (kvbase + j0 + jj) * KD + h * KHD;
            acc0 += pj * vrow[2 * lane];
            acc1 += pj * vrow[2 * lane + 1];
        }
    }
    float inv = 1.0f / lrun;
    float* orow = O + ((size_t)(b * qlen + qi)) * KD + h * KHD;
    orow[2 * lane]     = acc0 * inv;
    orow[2 * lane + 1] = acc1 * inv;
}

// =====================================================================
// Embedding + positional add (no LN)
// =====================================================================
__global__ __launch_bounds__(256) void embed_add_k(
    const int* __restrict__ ids, const float* __restrict__ emb,
    const float* __restrict__ pos, float* __restrict__ out,
    int rows, int seqlen)
{
    int i = blockIdx.x * blockDim.x + threadIdx.x;
    if (i >= rows * KD) return;
    int row = i >> 9;
    int d   = i & (KD - 1);
    int tok = ids[row];
    int p   = row % seqlen;
    out[i] = emb[(size_t)tok * KD + d] + pos[(size_t)p * KD + d];
}

// =====================================================================
// LayerNorm over D=512; one block per row, 256 threads x 2 elements.
// =====================================================================
__global__ __launch_bounds__(256) void layernorm_k(
    const float* __restrict__ X, float* __restrict__ Y,
    const float* __restrict__ sc, const float* __restrict__ bi)
{
    __shared__ float red[256];
    const int row = blockIdx.x;
    const int t   = threadIdx.x;
    const float* xr = X + (size_t)row * KD;
    float x0 = xr[t], x1 = xr[t + 256];
    red[t] = x0 + x1;
    __syncthreads();
    for (int o = 128; o > 0; o >>= 1) { if (t < o) red[t] += red[t + o]; __syncthreads(); }
    float mu = red[0] * (1.f / (float)KD);
    __syncthreads();
    float d0 = x0 - mu, d1 = x1 - mu;
    red[t] = d0 * d0 + d1 * d1;
    __syncthreads();
    for (int o = 128; o > 0; o >>= 1) { if (t < o) red[t] += red[t + o]; __syncthreads(); }
    float inv = rsqrtf(red[0] * (1.f / (float)KD) + 1e-6f);
    float* yr = Y + (size_t)row * KD;
    yr[t]       = d0 * inv * sc[t]       + bi[t];
    yr[t + 256] = d1 * inv * sc[t + 256] + bi[t + 256];
}

// ---------------------------------------------------------------------
// Host-side plumbing (assumes JAX pytree sorted-key leaf order)
// ---------------------------------------------------------------------
struct Dense { const float* b; const float* w; bf16_t* wt; };
struct LnP   { const float* b; const float* s; };
struct DecL  { Dense ck, co, cq, cv, fc1, fc2; LnP ln1, ln2, ln3; Dense sk, so, sq, sv; };
struct EncL  { Dense fc1, fc2, k_loc, k_sent; LnP ln1, ln2; Dense out, q_loc, q_tok;
               const float* anchors; Dense v_loc, v_sent; };

static inline void gemm(hipStream_t st, const float* A, const float* A2,
                        const Dense& d, const float* Res,
                        float* C, int M, int N, int K, int act) {
    dim3 g((N + 63) / 64, (M + 63) / 64);
    gemm_bf16_wmma<<<g, 256, 0, st>>>(A, A2, d.wt, d.b, Res, C, M, N, K, act);
}
static inline void attn(hipStream_t st, const float* Q, const float* Kp,
                        const float* Vp, float* O, const int* mask,
                        int nb, int qlen, int klen, int kvbs, int window, int causal) {
    int tasks = nb * KH * qlen;
    attn_wave<<<(tasks + 7) / 8, 256, 0, st>>>(Q, Kp, Vp, O, mask, nb, qlen, klen, kvbs, window, causal);
}
static inline void lnorm(hipStream_t st, const float* X, float* Y, const LnP& p, int rows) {
    layernorm_k<<<rows, 256, 0, st>>>(X, Y, p.s, p.b);
}

extern "C" void kernel_launch(void* const* d_in, const int* in_sizes, int n_in,
                              void* d_out, int out_size, void* d_ws, size_t ws_size,
                              hipStream_t stream) {
    const int* ids   = (const int*)d_in[0];   // input_ids [4,1024]
    const int* amask = (const int*)d_in[1];   // attention_mask [4,1024]
    const int* dids  = (const int*)d_in[2];   // decoder_input_ids [4,256]

    // ---- params: leaves in sorted-key pytree order ----
    int pi = 3;
    auto nf  = [&]() { return (const float*)d_in[pi++]; };
    auto rdD = [&](Dense& d) { d.b = nf(); d.w = nf(); d.wt = nullptr; };
    auto rdL = [&](LnP& l)   { l.b = nf(); l.s = nf(); };

    DecL dec[6];
    for (int i = 0; i < 6; ++i) {
        rdD(dec[i].ck); rdD(dec[i].co); rdD(dec[i].cq); rdD(dec[i].cv);
        rdD(dec[i].fc1); rdD(dec[i].fc2);
        rdL(dec[i].ln1); rdL(dec[i].ln2); rdL(dec[i].ln3);
        rdD(dec[i].sk); rdD(dec[i].so); rdD(dec[i].sq); rdD(dec[i].sv);
    }
    const float* embed = nf();
    EncL enc[6];
    for (int i = 0; i < 6; ++i) {
        rdD(enc[i].fc1); rdD(enc[i].fc2);
        rdD(enc[i].k_loc); rdD(enc[i].k_sent);
        rdL(enc[i].ln1); rdL(enc[i].ln2);
        rdD(enc[i].out); rdD(enc[i].q_loc); rdD(enc[i].q_tok);
        enc[i].anchors = nf();
        rdD(enc[i].v_loc); rdD(enc[i].v_sent);
    }
    Dense lm;  rdD(lm);
    LnP ln0, ln_enc, ln_f;
    rdL(ln0); rdL(ln_enc); rdL(ln_f);
    const float* pos = nf();

    // ---- workspace: f32 activations then bf16 weight arena ----
    float* ws = (float*)d_ws;
    size_t off = 0;
    auto alloc = [&](size_t n) { float* p = ws + off; off += n; return p; };
    const size_t MENC = (size_t)KB_ * KSRC;   // 4096 token rows
    float* XE = alloc(MENC * KD);
    float* HB = alloc(MENC * KD);
    float* Qb = alloc(MENC * KD);
    float* Kb = alloc(MENC * KD);
    float* Vb = alloc(MENC * KD);
    float* CA = alloc(MENC * KD);
    float* CL = alloc(MENC * KD);
    float* FF = alloc(MENC * (size_t)KFF);
    float* KS = alloc((size_t)KASENT * KD);
    float* VS = alloc((size_t)KASENT * KD);
    (void)ws_size; (void)in_sizes; (void)n_in; (void)out_size;

    bf16_t* warena = (bf16_t*)(ws + off);
    size_t woff = 0;
    auto prep = [&](Dense& d, int Kd_, int Nd_) {
        d.wt = warena + woff;
        woff += (size_t)Kd_ * Nd_;
        dim3 g((Nd_ + 31) / 32, (Kd_ + 31) / 32);
        convT_k<<<g, 256, 0, stream>>>(d.w, d.wt, Kd_, Nd_);
    };
    for (int i = 0; i < 6; ++i) {
        prep(enc[i].q_tok, KD, KD);  prep(enc[i].k_sent, KD, KD); prep(enc[i].v_sent, KD, KD);
        prep(enc[i].q_loc, KD, KD);  prep(enc[i].k_loc, KD, KD);  prep(enc[i].v_loc, KD, KD);
        prep(enc[i].out, KD, KD);    prep(enc[i].fc1, KD, KFF);   prep(enc[i].fc2, KFF, KD);
    }
    for (int i = 0; i < 6; ++i) {
        prep(dec[i].sq, KD, KD); prep(dec[i].sk, KD, KD); prep(dec[i].sv, KD, KD);
        prep(dec[i].so, KD, KD); prep(dec[i].cq, KD, KD); prep(dec[i].ck, KD, KD);
        prep(dec[i].cv, KD, KD); prep(dec[i].co, KD, KD);
        prep(dec[i].fc1, KD, KFF); prep(dec[i].fc2, KFF, KD);
    }
    prep(lm, KD, KV);

    // ================= ENCODER =================
    {
        int total = (int)(MENC * KD);
        embed_add_k<<<(total + 255) / 256, 256, 0, stream>>>(ids, embed, pos, XE, (int)MENC, KSRC);
    }
    lnorm(stream, XE, XE, ln0, (int)MENC);

    for (int l = 0; l < 6; ++l) {
        EncL& L = enc[l];
        lnorm(stream, XE, HB, L.ln1, (int)MENC);
        gemm(stream, L.anchors, nullptr, L.k_sent, nullptr, KS, KASENT, KD, KD, 0);
        gemm(stream, L.anchors, nullptr, L.v_sent, nullptr, VS, KASENT, KD, KD, 0);
        gemm(stream, HB, nullptr, L.q_tok, nullptr, Qb, (int)MENC, KD, KD, 0);
        attn(stream, Qb, KS, VS, CA, nullptr, KB_, KSRC, KASENT, 0, -1, 0);
        gemm(stream, HB, nullptr, L.q_loc, nullptr, Qb, (int)MENC, KD, KD, 0);
        gemm(stream, HB, nullptr, L.k_loc, nullptr, Kb, (int)MENC, KD, KD, 0);
        gemm(stream, HB, nullptr, L.v_loc, nullptr, Vb, (int)MENC, KD, KD, 0);
        attn(stream, Qb, Kb, Vb, CL, amask, KB_, KSRC, KSRC, KSRC, 128, 0);
        gemm(stream, CL, CA, L.out, XE, XE, (int)MENC, KD, KD, 0);
        lnorm(stream, XE, HB, L.ln2, (int)MENC);
        gemm(stream, HB, nullptr, L.fc1, nullptr, FF, (int)MENC, KFF, KD, 1);
        gemm(stream, FF, nullptr, L.fc2, XE, XE, (int)MENC, KD, KFF, 0);
    }
    lnorm(stream, XE, XE, ln_enc, (int)MENC);   // XE = encoder output

    // ================= DECODER =================
    const size_t MDEC = (size_t)KB_ * KTGT;     // 1024 token rows
    float* YD = CL;
    {
        int total = (int)(MDEC * KD);
        embed_add_k<<<(total + 255) / 256, 256, 0, stream>>>(dids, embed, pos, YD, (int)MDEC, KTGT);
    }
    for (int l = 0; l < 6; ++l) {
        DecL& L = dec[l];
        lnorm(stream, YD, HB, L.ln1, (int)MDEC);
        gemm(stream, HB, nullptr, L.sq, nullptr, Qb, (int)MDEC, KD, KD, 0);
        gemm(stream, HB, nullptr, L.sk, nullptr, Kb, (int)MDEC, KD, KD, 0);
        gemm(stream, HB, nullptr, L.sv, nullptr, Vb, (int)MDEC, KD, KD, 0);
        attn(stream, Qb, Kb, Vb, CA, nullptr, KB_, KTGT, KTGT, KTGT, -1, 1);
        gemm(stream, CA, nullptr, L.so, YD, YD, (int)MDEC, KD, KD, 0);
        lnorm(stream, YD, HB, L.ln2, (int)MDEC);
        gemm(stream, HB, nullptr, L.cq, nullptr, Qb, (int)MDEC, KD, KD, 0);
        gemm(stream, XE, nullptr, L.ck, nullptr, Kb, (int)MENC, KD, KD, 0);
        gemm(stream, XE, nullptr, L.cv, nullptr, Vb, (int)MENC, KD, KD, 0);
        attn(stream, Qb, Kb, Vb, CA, amask, KB_, KTGT, KSRC, KSRC, -1, 0);
        gemm(stream, CA, nullptr, L.co, YD, YD, (int)MDEC, KD, KD, 0);
        lnorm(stream, YD, HB, L.ln3, (int)MDEC);
        gemm(stream, HB, nullptr, L.fc1, nullptr, FF, (int)MDEC, KFF, KD, 1);
        gemm(stream, FF, nullptr, L.fc2, YD, YD, (int)MDEC, KD, KFF, 0);
    }
    lnorm(stream, YD, HB, ln_f, (int)MDEC);
    gemm(stream, HB, nullptr, lm, nullptr, (float*)d_out, (int)MDEC, KV, KD, 0);
}
